// VCALayer_2465311228098
// MI455X (gfx1250) — matrix operand order repaired and checked by reading, not compiled, not run on hardware
//
#include <hip/hip_runtime.h>
#include <cstdint>

// Problem constants (match reference)
#define QD 1280
#define CD 768
#define HH 8
#define NN 2
#define ZZ 2
#define RR 8
#define BFB 16
#define SS 4096
#define DH (QD / HH)     // 160
#define TT (NN * ZZ)     // 4
#define MROWS (BFB * SS) // 65536

// Set to 0 if the assembler rejects global_load_async_to_lds_b128.
#ifndef USE_ASYNC_LDS
#define USE_ASYNC_LDS 1
#endif

typedef __attribute__((ext_vector_type(16))) __bf16 v16bf;
typedef __attribute__((ext_vector_type(8)))  float  v8f;

// ---------------------------------------------------------------------------
// f32 -> bf16 conversion (grid-stride, 4 elems/thread/iter; all sizes %4==0)
// ---------------------------------------------------------------------------
__global__ __launch_bounds__(256) void cvt_f32_bf16(const float* __restrict__ in,
                                                    __bf16* __restrict__ out,
                                                    long n) {
  long i = (long)blockIdx.x * blockDim.x + threadIdx.x;
  long stride = (long)gridDim.x * blockDim.x;
  for (long k = i * 4; k < n; k += stride * 4) {
    float4 v = *(const float4*)(in + k);
    out[k + 0] = (__bf16)v.x;
    out[k + 1] = (__bf16)v.y;
    out[k + 2] = (__bf16)v.z;
    out[k + 3] = (__bf16)v.w;
  }
}

// ---------------------------------------------------------------------------
// K/V build: ctx3d = context + depth_pe; K = lora(ctx,Wk,Ak,Bk); V likewise.
// One block per (batch, token): 64 blocks total — negligible cost.
// ---------------------------------------------------------------------------
__global__ __launch_bounds__(256) void kv_build(
    const float* __restrict__ context, const float* __restrict__ depth_pe,
    const float* __restrict__ Wk, const float* __restrict__ Ak, const float* __restrict__ Bk,
    const float* __restrict__ Wv, const float* __restrict__ Av, const float* __restrict__ Bv,
    float* __restrict__ Kout, float* __restrict__ Vout) {
  __shared__ float ctx[CD];
  const int b = blockIdx.x / TT;
  const int t = blockIdx.x % TT;
  const int n = t / ZZ, z = t % ZZ;
  for (int c = threadIdx.x; c < CD; c += blockDim.x)
    ctx[c] = context[(size_t)(b * NN + n) * CD + c] + depth_pe[z * CD + c];
  __syncthreads();

  float ta[RR], tv[RR];
  for (int r = 0; r < RR; ++r) {
    float sk = 0.f, sv = 0.f;
    for (int c = 0; c < CD; ++c) {
      sk += ctx[c] * Ak[r * CD + c];
      sv += ctx[c] * Av[r * CD + c];
    }
    ta[r] = sk; tv[r] = sv;
  }
  for (int o = threadIdx.x; o < QD; o += blockDim.x) {
    float sk = 0.f, sv = 0.f;
    for (int c = 0; c < CD; ++c) {
      sk += ctx[c] * Wk[(size_t)o * CD + c];
      sv += ctx[c] * Wv[(size_t)o * CD + c];
    }
    for (int r = 0; r < RR; ++r) {
      sk += ta[r] * Bk[o * RR + r];
      sv += tv[r] * Bv[o * RR + r];
    }
    Kout[(size_t)(b * TT + t) * QD + o] = sk;
    Vout[(size_t)(b * TT + t) * QD + o] = sv;
  }
}

// ---------------------------------------------------------------------------
// bf16 WMMA GEMM: C[M,N] = A[M,K] @ B[N,K]^T (+ residual + bias epilogue)
// Block = 256 threads = 8 waves; tile 128x128, BK=32, double-buffered LDS.
// Staging uses gfx1250 async global->LDS (ASYNCcnt) when available.
// ---------------------------------------------------------------------------
#define BM 128
#define BN 128
#define BK 32
#define LSTR 40  // padded LDS row stride (bf16 elems): 80B, 16B-aligned

__device__ __forceinline__ v8f wmma_bf16(v16bf a, v16bf b, v8f c) {
  // (neg_a, A, neg_b, B, c_mod, C, reuse_a, reuse_b)
  return __builtin_amdgcn_wmma_f32_16x16x32_bf16(false, a, false, b, (short)0, c,
                                                 false, false);
}

// Low 32 bits of a generic pointer to LDS == AS(3) byte offset (generic LDS
// pointers are {shared_aperture_hi32, lds_offset32}; hardware truncates).
__device__ __forceinline__ unsigned lds_lo32(const void* p) {
  return (unsigned)(uintptr_t)p;
}

__device__ __forceinline__ void async_ld_b128(unsigned lds, const void* g) {
  unsigned long long ga = (unsigned long long)(uintptr_t)g;
  asm volatile("global_load_async_to_lds_b128 %0, %1, off"
               :: "v"(lds), "v"(ga) : "memory");
}

__device__ __forceinline__ void wait_asynccnt0() {
  asm volatile("s_wait_asynccnt 0x0" ::: "memory");
}

__global__ __launch_bounds__(256) void gemm_bf16(
    const __bf16* __restrict__ A, const __bf16* __restrict__ B,
    float* __restrict__ C, const float* __restrict__ residual,
    const float* __restrict__ bias, int M, int Nn, int K) {
  __shared__ alignas(16) __bf16 As[2][BM][LSTR];
  __shared__ alignas(16) __bf16 Bs[2][BN][LSTR];

  const int tid   = threadIdx.x;
  const int lane  = tid & 31;
  const int wid   = tid >> 5;
  const int waveM = wid & 3;   // 0..3 -> 32-row slab
  const int waveN = wid >> 2;  // 0..1 -> 64-col slab
  const int m0 = blockIdx.y * BM;
  const int n0 = blockIdx.x * BN;

  // cooperative tile loader: 2 threads per row, 32B (two b128 chunks) each
  const int ldRow = tid >> 1;          // 0..127
  const int ldOff = (tid & 1) * 16;    // 0 or 16 elems
  const __bf16* gA = A + (size_t)(m0 + ldRow) * K + ldOff;
  const __bf16* gB = B + (size_t)(n0 + ldRow) * K + ldOff;

  unsigned ldsA[2], ldsB[2];
  ldsA[0] = lds_lo32(&As[0][ldRow][ldOff]);
  ldsA[1] = lds_lo32(&As[1][ldRow][ldOff]);
  ldsB[0] = lds_lo32(&Bs[0][ldRow][ldOff]);
  ldsB[1] = lds_lo32(&Bs[1][ldRow][ldOff]);

  v8f acc[2][4];
#pragma unroll
  for (int i = 0; i < 2; ++i)
#pragma unroll
    for (int j = 0; j < 4; ++j)
#pragma unroll
      for (int r = 0; r < 8; ++r) acc[i][j][r] = 0.0f;

  const int KT = K / BK;  // 40

  // prologue: stage 0
#if USE_ASYNC_LDS
  async_ld_b128(ldsA[0], gA);
  async_ld_b128(ldsA[0] + 16, gA + 8);
  async_ld_b128(ldsB[0], gB);
  async_ld_b128(ldsB[0] + 16, gB + 8);
  wait_asynccnt0();
#else
  {
    uint4 alo = *(const uint4*)(gA);
    uint4 ahi = *(const uint4*)(gA + 8);
    uint4 blo = *(const uint4*)(gB);
    uint4 bhi = *(const uint4*)(gB + 8);
    *(uint4*)(&As[0][ldRow][ldOff])     = alo;
    *(uint4*)(&As[0][ldRow][ldOff + 8]) = ahi;
    *(uint4*)(&Bs[0][ldRow][ldOff])     = blo;
    *(uint4*)(&Bs[0][ldRow][ldOff + 8]) = bhi;
  }
#endif
  __syncthreads();

  const int mlane = lane & 15;
  const int kloA  = (lane < 16) ? 0 : 8;   // unused; A frag uses kbB split (k = 16*half + e)
  const int kbB   = (lane < 16) ? 0 : 16;  // B frag: lane-half K split

  for (int kt = 0; kt < KT; ++kt) {
    const int cur = kt & 1;
    const int nxt = cur ^ 1;
    if (kt + 1 < KT) {
      const __bf16* ga = gA + (size_t)(kt + 1) * BK;
      const __bf16* gb = gB + (size_t)(kt + 1) * BK;
#if USE_ASYNC_LDS
      async_ld_b128(ldsA[nxt], ga);
      async_ld_b128(ldsA[nxt] + 16, ga + 8);
      async_ld_b128(ldsB[nxt], gb);
      async_ld_b128(ldsB[nxt] + 16, gb + 8);
#else
      uint4 alo = *(const uint4*)(ga);
      uint4 ahi = *(const uint4*)(ga + 8);
      uint4 blo = *(const uint4*)(gb);
      uint4 bhi = *(const uint4*)(gb + 8);
      *(uint4*)(&As[nxt][ldRow][ldOff])     = alo;
      *(uint4*)(&As[nxt][ldRow][ldOff + 8]) = ahi;
      *(uint4*)(&Bs[nxt][ldRow][ldOff])     = blo;
      *(uint4*)(&Bs[nxt][ldRow][ldOff + 8]) = bhi;
#endif
      if (kt + 2 < KT) {  // global_prefetch_b8 for the stage after next
        __builtin_prefetch(gA + (size_t)(kt + 2) * BK, 0, 1);
        __builtin_prefetch(gB + (size_t)(kt + 2) * BK, 0, 1);
      }
    }

    // Load ALL fragments into distinct registers first, so the ds_loads form
    // one clause with a single wait, then the 8 WMMAs issue back-to-back.
    v16bf afrag[2], bfrag[4];
#pragma unroll
    for (int ti = 0; ti < 2; ++ti) {
      const int mr = waveM * 32 + ti * 16 + mlane;
#pragma unroll
      for (int e = 0; e < 16; ++e) {
        const int kk = kbB + e;  // K split per lane-half
        afrag[ti][e] = As[cur][mr][kk];
      }
    }
#pragma unroll
    for (int tj = 0; tj < 4; ++tj) {
      const int nr = waveN * 64 + tj * 16 + mlane;
#pragma unroll
      for (int e = 0; e < 16; ++e) bfrag[tj][e] = Bs[cur][nr][kbB + e];
    }
#pragma unroll
    for (int tj = 0; tj < 4; ++tj) {
      acc[0][tj] = wmma_bf16(afrag[0], bfrag[tj], acc[0][tj]);
      acc[1][tj] = wmma_bf16(afrag[1], bfrag[tj], acc[1][tj]);
    }

#if USE_ASYNC_LDS
    if (kt + 1 < KT) wait_asynccnt0();  // stage(nxt) resident before flip
#endif
    __syncthreads();
  }

  // epilogue: D layout — VGPR r: lanes<16 row r, lanes>=16 row r+8
  const int rAdd = (lane < 16) ? 0 : 8;
#pragma unroll
  for (int ti = 0; ti < 2; ++ti) {
#pragma unroll
    for (int tj = 0; tj < 4; ++tj) {
#pragma unroll
      for (int r = 0; r < 8; ++r) {
        const int row = m0 + waveM * 32 + ti * 16 + r + rAdd;
        const int col = n0 + waveN * 64 + tj * 16 + mlane;
        const size_t idx = (size_t)row * Nn + col;
        float v = acc[ti][tj][r];
        if (bias) v += bias[col];
        if (residual) v += residual[idx];
        C[idx] = v;
      }
    }
  }
}

// ---------------------------------------------------------------------------
// Attention over 4 context tokens: one query row per wave (wave32).
// ---------------------------------------------------------------------------
__global__ __launch_bounds__(256) void attn_kernel(
    const float* __restrict__ Q, const float* __restrict__ Kc,
    const float* __restrict__ Vc, __bf16* __restrict__ attn_out) {
  __shared__ float Ks[TT * QD];
  __shared__ float Vs[TT * QD];
  const int tid  = threadIdx.x;
  const int lane = tid & 31;
  const int wid  = tid >> 5;
  const int row  = blockIdx.x * 8 + wid;
  const int b    = row >> 12;  // 4096 rows per batch; block rows share a batch

  for (int i = tid; i < TT * QD; i += 256) {
    Ks[i] = Kc[(size_t)b * TT * QD + i];
    Vs[i] = Vc[(size_t)b * TT * QD + i];
  }
  __syncthreads();

  float q[40];
#pragma unroll
  for (int i = 0; i < 40; ++i) q[i] = Q[(size_t)row * QD + i * 32 + lane];

  const float scale = 0.07905694150420949f;  // 1/sqrt(160)
  float sig[HH][TT];
#pragma unroll
  for (int h = 0; h < HH; ++h) {
#pragma unroll
    for (int t = 0; t < TT; ++t) {
      float p = 0.f;
#pragma unroll
      for (int j = 0; j < 5; ++j) {  // head h covers lane-chunks 5h..5h+4
        const int i = h * 5 + j;
        p += q[i] * Ks[t * QD + i * 32 + lane];
      }
      p += __shfl_xor(p, 16);
      p += __shfl_xor(p, 8);
      p += __shfl_xor(p, 4);
      p += __shfl_xor(p, 2);
      p += __shfl_xor(p, 1);
      sig[h][t] = 1.0f / (1.0f + __expf(-p * scale));
    }
  }

  float sm[TT];
#pragma unroll
  for (int t = 0; t < TT; ++t) {
    float s = 0.f;
#pragma unroll
    for (int h = 0; h < HH; ++h) s += sig[h][t];
    sm[t] = s * 0.125f;
  }
  const float op0 = fminf(sm[0] + sm[2], 1.0f);  // z=0: t = n*ZZ + 0 -> {0,2}
  const float Tz[2] = {1.0f, 1.0f - op0};

  float w[HH][TT];
#pragma unroll
  for (int h = 0; h < HH; ++h)
#pragma unroll
    for (int t = 0; t < TT; ++t) w[h][t] = Tz[t % ZZ] * sig[h][t];

#pragma unroll
  for (int i = 0; i < 40; ++i) {
    const int d = i * 32 + lane;
    const int h = d / DH;
    float o = 0.f;
#pragma unroll
    for (int t = 0; t < TT; ++t) o += w[h][t] * Vs[t * QD + d];
    attn_out[(size_t)row * QD + d] = (__bf16)o;
  }
}

// ---------------------------------------------------------------------------
// Launch
// ---------------------------------------------------------------------------
extern "C" void kernel_launch(void* const* d_in, const int* in_sizes, int n_in,
                              void* d_out, int out_size, void* d_ws, size_t ws_size,
                              hipStream_t stream) {
  const float* x        = (const float*)d_in[0];
  const float* context  = (const float*)d_in[1];
  const float* Wq       = (const float*)d_in[2];
  const float* Wk       = (const float*)d_in[3];
  const float* Ak       = (const float*)d_in[4];
  const float* Bk       = (const float*)d_in[5];
  const float* Wv       = (const float*)d_in[6];
  const float* Av       = (const float*)d_in[7];
  const float* Bv       = (const float*)d_in[8];
  const float* depth_pe = (const float*)d_in[9];
  const float* Wout     = (const float*)d_in[10];
  const float* bout     = (const float*)d_in[11];
  float* out = (float*)d_out;

  char* ws = (char*)d_ws;
  size_t off = 0;
  auto carve = [&](size_t bytes) -> void* {
    void* p = ws + off;
    off += (bytes + 255) & ~(size_t)255;
    return p;
  };
  __bf16* x_bf    = (__bf16*)carve((size_t)MROWS * QD * sizeof(__bf16));
  __bf16* Wq_bf   = (__bf16*)carve((size_t)QD * QD * sizeof(__bf16));
  __bf16* Wo_bf   = (__bf16*)carve((size_t)QD * QD * sizeof(__bf16));
  float*  Qbuf    = (float*) carve((size_t)MROWS * QD * sizeof(float));
  __bf16* attn_bf = (__bf16*)carve((size_t)MROWS * QD * sizeof(__bf16));
  float*  Kbuf    = (float*) carve((size_t)BFB * TT * QD * sizeof(float));
  float*  Vbuf    = (float*) carve((size_t)BFB * TT * QD * sizeof(float));

  // 1) precision staging: f32 -> bf16 for GEMM operands
  cvt_f32_bf16<<<4096, 256, 0, stream>>>(x, x_bf, (long)MROWS * QD);
  cvt_f32_bf16<<<512, 256, 0, stream>>>(Wq, Wq_bf, (long)QD * QD);
  cvt_f32_bf16<<<512, 256, 0, stream>>>(Wout, Wo_bf, (long)QD * QD);

  // 2) K/V via LoRA on depth-expanded context (tiny)
  kv_build<<<BFB * TT, 256, 0, stream>>>(context, depth_pe, Wk, Ak, Bk,
                                         Wv, Av, Bv, Kbuf, Vbuf);

  // 3) Q = x @ Wq^T  (WMMA bf16, f32 accumulate)
  dim3 ggrid(QD / BN, MROWS / BM);  // 10 x 512
  gemm_bf16<<<ggrid, 256, 0, stream>>>(x_bf, Wq_bf, Qbuf, nullptr, nullptr,
                                       MROWS, QD, QD);

  // 4) volumetric attention over 4 context tokens -> attn (bf16)
  attn_kernel<<<MROWS / 8, 256, 0, stream>>>(Qbuf, Kbuf, Vbuf, attn_bf);

  // 5) out = x + attn @ Wout^T + bout  (WMMA bf16 with fused epilogue)
  gemm_bf16<<<ggrid, 256, 0, stream>>>(attn_bf, Wo_bf, out, x, bout,
                                       MROWS, QD, QD);
}
